// HyperConv2d_56152402428196
// MI455X (gfx1250) — compile-verified, matrix operand order checked by reading, not run on hardware
//
#include <hip/hip_runtime.h>
#include <cstdint>
#include <cstddef>

// ---------------------------------------------------------------------------
// HyperConv2d for MI455X (gfx1250, wave32, WMMA + TDM)
// B=16, CIN=COUT=64, K=3, H=W=128, HE=8, PAD=1
// Phase 1: hypernetwork -> per-sample f16 weights in workspace, layout
//          [b][kk=kh*3+kw][cout][cin]  (contiguous cin for A-fragments)
// Phase 2: implicit-GEMM conv, one block per (b, output row),
//          TENSOR_LOAD_TO_LDS weight staging + v_wmma_f32_16x16x32_f16.
// ---------------------------------------------------------------------------

typedef __attribute__((ext_vector_type(16))) _Float16 v16h;
typedef __attribute__((ext_vector_type(8)))  float    v8f;
typedef __attribute__((ext_vector_type(4)))  unsigned int u32x4;
typedef __attribute__((ext_vector_type(4)))  int      i32x4;
typedef __attribute__((ext_vector_type(8)))  int      i32x8;

union FragH {            // 16 f16 = 32 bytes = one WMMA A/B operand (8 VGPRs)
    v16h  v;
    _Float16 h[16];
    uint4 q[2];          // two 16-byte LDS vector loads
};

#define NB    16
#define NCIN  64
#define NCOUT 64
#define NH    128
#define NW    128
#define NHE   8
#define XCOLS 130        // 128 + left/right halo (pad=1)
#define WELEMS (9 * NCOUT * NCIN)       // 36864 f16 per sample

#if __has_builtin(__builtin_amdgcn_tensor_load_to_lds)
#define HAVE_TDM 1
#else
#define HAVE_TDM 0
#endif

__device__ __forceinline__ float lrelu(float v) { return v > 0.f ? v : 0.01f * v; }

// ---------------------------------------------------------------------------
// Phase 1: hypernetwork. One block per sample b. 256 threads.
// ---------------------------------------------------------------------------
__global__ __launch_bounds__(256) void hyper_weights_kernel(
    const float* __restrict__ hvec,  const float* __restrict__ fc_w,
    const float* __restrict__ fc_b,
    const float* __restrict__ w1, const float* __restrict__ b1,
    const float* __restrict__ w2, const float* __restrict__ b2,
    const float* __restrict__ w3, const float* __restrict__ b3,
    _Float16* __restrict__ wout)     // [NB][9][NCOUT][NCIN]
{
    __shared__ float sk0[16 * 9];    // [c][kl]
    __shared__ float sk1[32 * 9];
    __shared__ float sk2[32 * 9];

    const int b   = blockIdx.x;
    const int tid = threadIdx.x;

    if (tid < 16 * 9) {
        float acc = fc_b[tid];
        #pragma unroll
        for (int e = 0; e < NHE; ++e) acc += hvec[b * NHE + e] * fc_w[tid * NHE + e];
        sk0[tid] = lrelu(acc);       // j = c*9 + kl  (reshape [16,3,3])
    }
    __syncthreads();

    if (tid < 32 * 9) {
        const int o = tid / 9, kl = tid % 9;
        float acc = b1[o];
        #pragma unroll
        for (int c = 0; c < 16; ++c) acc += w1[o * 16 + c] * sk0[c * 9 + kl];
        sk1[tid] = lrelu(acc);
    }
    __syncthreads();

    if (tid < 32 * 9) {
        const int o = tid / 9, kl = tid % 9;
        float acc = b2[o];
        #pragma unroll
        for (int c = 0; c < 32; ++c) acc += w2[o * 32 + c] * sk1[c * 9 + kl];
        sk2[tid] = lrelu(acc);
    }
    __syncthreads();

    _Float16* wb = wout + (size_t)b * WELEMS;
    for (int t = tid; t < NCOUT * NCIN * 9; t += 256) {
        const int o  = t / 9;              // o = cout*64 + cin
        const int kl = t % 9;
        const int cout = o / NCIN;
        const int cin  = o % NCIN;
        float acc = b3[o];
        #pragma unroll
        for (int c = 0; c < 32; ++c) acc += w3[o * 32 + c] * sk2[c * 9 + kl];
        wb[(kl * NCOUT + cout) * NCIN + cin] = (_Float16)acc;
    }
}

// ---------------------------------------------------------------------------
// Phase 2: implicit-GEMM conv via WMMA, weights staged by the TDM.
// Grid: NB*NH blocks (one output row of one sample). 256 threads = 8 waves.
// Dynamic LDS: 72KB weights + 48.75KB im2col halo (320KB WGP LDS).
// ---------------------------------------------------------------------------
__global__ __launch_bounds__(256) void hyperconv_wmma_kernel(
    const float* __restrict__ x,        // [NB][NCIN][NH][NW] f32
    const _Float16* __restrict__ wall,  // [NB][9][NCOUT][NCIN] f16
    const float* __restrict__ bias,     // [NCOUT]
    float* __restrict__ out)            // [NB][NCOUT][NH][NW] f32
{
    extern __shared__ _Float16 smem[];
    _Float16* wlds = smem;                        // 9*64*64   = 36864 f16
    _Float16* xs   = smem + WELEMS;               // 3*130*64  = 24960 f16, [r][col][cin]

    const int tid  = threadIdx.x;
    const int b    = blockIdx.x >> 7;
    const int hrow = blockIdx.x & (NH - 1);
    const _Float16* wsrc = wall + (size_t)b * WELEMS;

#if HAVE_TDM
    // ---- TDM: DMA the 73,728-byte per-sample weight block into LDS ----
    // D# per CDNA5 ISA: 1 descriptor, data_size=8B,
    // single row tile of 9216 elements, dims 2..4 degenerate.
    if (tid < 32) {   // wave 0 only (wave-uniform branch; tensor ops ignore EXEC)
        const unsigned long long ga = (unsigned long long)(uintptr_t)wsrc;
        const unsigned lds_off = (unsigned)(uintptr_t)wlds;  // low 32b = LDS offset
        const unsigned NE = WELEMS * 2 / 8;                  // 9216 x 8-byte elems

        u32x4 g0 = { 0x1u,                       // count=1, user mode
                     lds_off,                    // lds_addr
                     (unsigned)(ga & 0xffffffffu),
                     (unsigned)(ga >> 32) | (2u << 30) };    // type=2 ("image")

        i32x8 g1 = { (int)(3u << 16),            // mask=0 | data_size=3 (8B)
                     (int)((NE & 0xffffu) << 16),// tensor_dim0[15:0]  (bits 63:48)
                     (int)((NE >> 16) | (1u << 16)), // td0[31:16] | tensor_dim1=1
                     (int)(NE << 16),            // td1[31:16]=0 | tile_dim0=9216
                     (int)1,                     // tile_dim1=1 | tile_dim2=0
                     (int)NE,                    // tensor_dim0_stride[31:0]
                     0,                          // stride0[47:32] | stride1[15:0]
                     0 };                        // stride1[47:16]

        i32x4 g2 = { 1, 1, 0, 0 };               // tensor_dim2=1, tensor_dim3=1
        i32x4 g3 = { 0, 0, 0, 0 };

#if __has_include(<hip/amd_detail/amd_gfx1250_TDM.h>)
        i32x8 g4 = { 0, 0, 0, 0, 0, 0, 0, 0 };
        __builtin_amdgcn_tensor_load_to_lds(g0, g1, g2, g3, g4, 0);
#else
        __builtin_amdgcn_tensor_load_to_lds(g0, g1, g2, g3, 0);
#endif
    }
#else
    // ---- fallback: manual 16B vector copies ----
    {
        const uint4* wg = reinterpret_cast<const uint4*>(wsrc);
        uint4* wl = reinterpret_cast<uint4*>(wlds);
        for (int i = tid; i < WELEMS / 8; i += 256) wl[i] = wg[i];
    }
#endif

    // ---- stage im2col halo: rows hrow-1..hrow+1, cols -1..128, f32->f16 ----
    // (runs on VALU while the TDM streams the weights)
    for (int i = tid; i < 3 * NCIN * XCOLS; i += 256) {
        const int col  = i % XCOLS;                 // 0..129 -> gw = col-1
        const int rest = i / XCOLS;
        const int cin  = rest % NCIN;
        const int r    = rest / NCIN;               // 0..2   -> gh = hrow-1+r
        const int gh = hrow - 1 + r;
        const int gw = col - 1;
        float v = 0.f;
        if ((unsigned)gh < (unsigned)NH && (unsigned)gw < (unsigned)NW)
            v = x[(((size_t)b * NCIN + cin) * NH + gh) * NW + gw];
        xs[(r * XCOLS + col) * NCIN + cin] = (_Float16)v;   // cin contiguous
    }

#if HAVE_TDM
    if (tid < 32) __builtin_amdgcn_s_wait_tensorcnt(0);  // issuing wave drains TDM
#endif
    __syncthreads();                                     // publish LDS to all waves

    const int wave  = tid >> 5;
    const int lane  = tid & 31;
    const int half  = (lane >> 4) & 1;   // K-halves per ISA 16-bit A/B layout
    const int hm    = lane & 15;         // M (A) / N (B) index within tile
    const int ncol0 = wave * 16;         // this wave's 16 output pixels

    v8f acc[4] = {};                     // 4 M-tiles of 16 couts each

    #pragma unroll
    for (int kk = 0; kk < 9; ++kk) {
        const int kh = kk / 3, kw = kk % 3;
        #pragma unroll
        for (int cb = 0; cb < NCIN; cb += 32) {   // K-chunk of 32 cin
            const int klo = cb + 8 * half;        // K = klo..klo+7   (VGPR0..3)
            const int khi = cb + 16 + 8 * half;   // K = khi..khi+7   (VGPR4..7)

            FragH fb;                              // B: x-patch [32 x 16]
            const _Float16* xrow = xs + (kh * XCOLS + (ncol0 + hm + kw)) * NCIN;
            fb.q[0] = *reinterpret_cast<const uint4*>(xrow + klo);
            fb.q[1] = *reinterpret_cast<const uint4*>(xrow + khi);

            #pragma unroll
            for (int mt = 0; mt < 4; ++mt) {
                FragH fa;                          // A: weights [16 x 32]
                const _Float16* wrow = wlds + (kk * NCOUT + mt * 16 + hm) * NCIN;
                fa.q[0] = *reinterpret_cast<const uint4*>(wrow + klo);
                fa.q[1] = *reinterpret_cast<const uint4*>(wrow + khi);

                acc[mt] = __builtin_amdgcn_wmma_f32_16x16x32_f16(
                    /*neg_a=*/false, fa.v, /*neg_b=*/false, fb.v,
                    /*c_mod=*/(short)0, acc[mt],
                    /*reuse_a=*/false, /*reuse_b=*/false);
            }
        }
    }

    // ---- epilogue: C/D layout -> global. VGPR r: lanes0-15 M=r, 16-31 M=r+8
    const int col = ncol0 + hm;
    #pragma unroll
    for (int mt = 0; mt < 4; ++mt) {
        #pragma unroll
        for (int r = 0; r < 8; ++r) {
            const int cout = mt * 16 + r + 8 * half;
            out[(((size_t)b * NCOUT + cout) * NH + hrow) * NW + col] =
                acc[mt][r] + bias[cout];
        }
    }
}

// ---------------------------------------------------------------------------
extern "C" void kernel_launch(void* const* d_in, const int* in_sizes, int n_in,
                              void* d_out, int out_size, void* d_ws, size_t ws_size,
                              hipStream_t stream) {
    const float* x    = (const float*)d_in[0];
    const float* h    = (const float*)d_in[1];
    const float* fc_w = (const float*)d_in[2];
    const float* fc_b = (const float*)d_in[3];
    const float* w1   = (const float*)d_in[4];
    const float* b1   = (const float*)d_in[5];
    const float* w2   = (const float*)d_in[6];
    const float* b2   = (const float*)d_in[7];
    const float* w3   = (const float*)d_in[8];
    const float* b3   = (const float*)d_in[9];
    const float* bias = (const float*)d_in[10];
    float* out = (float*)d_out;

    _Float16* wtmp = (_Float16*)d_ws;   // 16*9*64*64 f16 = 1.15 MB

    hyper_weights_kernel<<<NB, 256, 0, stream>>>(h, fc_w, fc_b, w1, b1, w2, b2,
                                                 w3, b3, wtmp);

    const size_t smem_bytes =
        (size_t)(WELEMS + 3 * XCOLS * NCIN) * sizeof(_Float16); // 123,648 B
    hyperconv_wmma_kernel<<<NB * NH, 256, smem_bytes, stream>>>(x, wtmp, bias, out);
}
